// C4D_36550171689493
// MI455X (gfx1250) — matrix operand order repaired and checked by reading, not compiled, not run on hardware
//
#include <hip/hip_runtime.h>
#include <hip/hip_bf16.h>

// ---------------------------------------------------------------------------
// Types for WMMA fragments / TDM descriptors (wave32, gfx1250)
// ---------------------------------------------------------------------------
typedef __attribute__((ext_vector_type(16))) __bf16          v16bf;
typedef __attribute__((ext_vector_type(8)))  float           v8f;
typedef __attribute__((ext_vector_type(16))) unsigned short  us16;
typedef __attribute__((ext_vector_type(8)))  unsigned short  us8;
typedef __attribute__((ext_vector_type(4)))  unsigned int    v4u;
typedef __attribute__((ext_vector_type(8)))  int             v8i_t;
typedef __attribute__((ext_vector_type(4)))  int             v4i_t;

__device__ __forceinline__ unsigned short f32_to_bf16_bits(float f) {
    unsigned int u = __float_as_uint(f);
    unsigned int lsb = (u >> 16) & 1u;
    u += 0x7fffu + lsb;                 // round-to-nearest-even
    return (unsigned short)(u >> 16);
}
__device__ __forceinline__ float bf16_bits_to_f32(unsigned short h) {
    return __uint_as_float(((unsigned int)h) << 16);
}

// ---------------------------------------------------------------------------
// Weight packing: f32 (Co, Ci, 3,3,3,3) -> bf16 [Co][Kpad], K = ci*81 + tap
// ---------------------------------------------------------------------------
__global__ void pack_w_bf16(const float* __restrict__ w, unsigned short* __restrict__ wp,
                            int Co, int Ktot, int Kpad) {
    size_t i = (size_t)blockIdx.x * blockDim.x + threadIdx.x;
    size_t total = (size_t)Co * Kpad;
    if (i >= total) return;
    int co = (int)(i / Kpad);
    int k  = (int)(i - (size_t)co * Kpad);
    float v = (k < Ktot) ? w[(size_t)co * Ktot + k] : 0.0f;
    wp[i] = f32_to_bf16_bits(v);
}

__global__ void cvt_f32_bf16(const float* __restrict__ x, unsigned short* __restrict__ y,
                             size_t n) {
    size_t i = (size_t)blockIdx.x * blockDim.x + threadIdx.x;
    if (i < n) y[i] = f32_to_bf16_bits(x[i]);
}

// ---------------------------------------------------------------------------
// Implicit-GEMM 4-D convolution, v_wmma_f32_16x16x32_bf16, LDS-staged tiles.
//   act : bf16 activations  [Ci][Li][D][H][W]
//   wp  : bf16 packed wts   [Co][Kpad]
//   out : f32               [Co][N],  N = Lo*D*H*W   (y = conv + bias)
// Block = 256 threads = 8 waves; block tile = 32 co x 64 n; wave tile 16x16.
// grid.x = ceil(N/64), grid.y = Co/32   (all Co are multiples of 32)
// A tile (32co x 32k, 2KB) loaded by the Tensor Data Mover; B tile
// (64n x 32k, 4KB) gathered cooperatively (im2col) with a precomputed tap LUT.
// ---------------------------------------------------------------------------
__global__ __launch_bounds__(256)
void conv4d_wmma_bf16(const unsigned short* __restrict__ act,
                      const unsigned short* __restrict__ wp,
                      const float* __restrict__ bias,
                      float* __restrict__ out,
                      int Ci, int Co, int Li, int Lo,
                      int D, int H, int W, int strideL, int Kpad) {
    __shared__ __align__(16) unsigned short aTile[32 * 32];   // [co][k]
    __shared__ __align__(16) unsigned short bTile[64 * 32];   // [n][k]
    __shared__ unsigned int tap[81];

    const int tid    = threadIdx.x;
    const int lane   = tid & 31;
    const int wave   = tid >> 5;     // 0..7
    const int co_sub = wave >> 2;    // 0..1
    const int n_sub  = wave & 3;     // 0..3
    const int ln     = lane & 15;
    const int kg     = lane >> 4;    // half-wave group

    const int N     = Lo * D * H * W;
    const int HW    = H * W;
    const int DHW   = D * HW;
    const int LiDHW = Li * DHW;
    const int co_blk = blockIdx.y * 32;

    // Constant tap LUT: t -> (kl,kd,kh,kw) packed in bytes
    if (tid < 81) {
        int t = tid;
        int kl = t / 27; t -= kl * 27;
        int kd = t / 9;  t -= kd * 9;
        int kh = t / 3;
        int kw = t - kh * 3;
        tap[tid] = (unsigned)kl | ((unsigned)kd << 8) | ((unsigned)kh << 16) |
                   ((unsigned)kw << 24);
    }

    // ---- B-fill ownership: thread fills bTile[n_fill][koff..koff+7] ----
    const int n_fill = tid >> 2;          // 0..63
    const int koff   = (tid & 3) * 8;     // 0,8,16,24
    const int gn     = blockIdx.x * 64 + n_fill;
    const bool gn_valid = (gn < N);
    int lo, dd, hh, ww;
    {
        int t = gn_valid ? gn : 0;
        ww = t % W; t /= W;
        hh = t % H; t /= H;
        dd = t % D; t /= D;
        lo = t;
    }
    const int lbase = lo * strideL - 1;
    const int dbase = dd - 1, hbase = hh - 1, wbase = ww - 1;

    // Incremental (ci, tap) tracking for k = kb + koff (koff < 81)
    int ci0 = 0, t0 = koff;

    // ---- TDM descriptor (constant part) for the A weight tile ----
    const unsigned long long wtile0 =
        (unsigned long long)(size_t)wp + (unsigned long long)(size_t)co_blk * Kpad * 2ull;
    const unsigned int lds_a = (unsigned int)(unsigned long long)(void*)&aTile[0];

    __syncthreads();   // tap LUT ready

    v8f acc = {};

    for (int kb = 0; kb < Kpad; kb += 32) {
        // ---------------- A tile: 32 rows x 32 k (64B rows, stride Kpad*2) ----
#if __has_builtin(__builtin_amdgcn_tensor_load_to_lds)
        if (wave == 0) {
            const unsigned long long ga = wtile0 + (unsigned long long)kb * 2ull;
            v4u g0;
            g0[0] = 1u;                                         // count=1
            g0[1] = lds_a;                                      // lds_addr
            g0[2] = (unsigned int)(ga & 0xffffffffull);         // global_addr[31:0]
            g0[3] = (unsigned int)((ga >> 32) & 0x1ffffffull) | (2u << 30); // addr[56:32]|type=2
            v8i_t g1;
            g1[0] = (int)(1u << 16);                            // data_size=1 (2 bytes)
            g1[1] = (int)(((unsigned)Kpad & 0xffffu) << 16);    // tensor_dim0[15:0]
            g1[2] = (int)((((unsigned)Co & 0xffffu) << 16) |    // tensor_dim1[15:0]
                          (((unsigned)Kpad >> 16) & 0xffffu));  // tensor_dim0[31:16]
            g1[3] = (int)((32u << 16) |                         // tile_dim0 = 32
                          (((unsigned)Co >> 16) & 0xffffu));    // tensor_dim1[31:16]
            g1[4] = 32;                                         // tile_dim1 = 32, tile_dim2=0
            g1[5] = Kpad;                                       // tensor_dim0_stride[31:0]
            g1[6] = 0;
            g1[7] = 0;
            v4i_t gz4 = {0, 0, 0, 0};
            v8i_t gz8 = {0, 0, 0, 0, 0, 0, 0, 0};
            __builtin_amdgcn_tensor_load_to_lds(g0, g1, gz4, gz4, gz8, 0);
        }
#else
        if (tid < 128) {
            const int arow = tid >> 2;            // 0..31
            const int ak   = (tid & 3) * 8;       // 0..24
            *(us8*)&aTile[arow * 32 + ak] =
                *(const us8*)(wp + (size_t)(co_blk + arow) * Kpad + kb + ak);
        }
#endif

        // ---------------- B tile: im2col gather (8 ushorts/thread) ----------
        us8 bv;
#pragma unroll
        for (int j = 0; j < 8; ++j) {
            int tj = t0 + j;
            int cij = ci0;
            if (tj >= 81) { tj -= 81; ++cij; }
            unsigned short v = 0;
            if (gn_valid && cij < Ci) {
                const unsigned u = tap[tj];
                const int li = lbase + (int)(u & 0xffu);
                const int di = dbase + (int)((u >> 8) & 0xffu);
                const int hi = hbase + (int)((u >> 16) & 0xffu);
                const int wi = wbase + (int)(u >> 24);
                if ((unsigned)li < (unsigned)Li && (unsigned)di < (unsigned)D &&
                    (unsigned)hi < (unsigned)H && (unsigned)wi < (unsigned)W) {
                    v = act[(size_t)cij * LiDHW + (size_t)(li * DHW + di * HW + hi * W + wi)];
                }
            }
            bv[j] = v;
        }
        *(us8*)&bTile[n_fill * 32 + koff] = bv;
        t0 += 32;
        if (t0 >= 81) { t0 -= 81; ++ci0; }

#if __has_builtin(__builtin_amdgcn_tensor_load_to_lds)
#if __has_builtin(__builtin_amdgcn_s_wait_tensorcnt)
        if (wave == 0) __builtin_amdgcn_s_wait_tensorcnt(0);
#endif
#endif
        __syncthreads();   // A + B tiles resident in LDS

        // -------- Fragments from LDS (contiguous 16B ds loads) --------------
        // A (16x32 bf16): lanes 0-15: halves0-7=K0-7, halves8-15=K16-23;
        //                 lanes 16-31: K8-15 / K24-31.  Row m = ln.
        us16 afrag;
        {
            const int arow = co_sub * 16 + ln;
            const us8 a_lo = *(const us8*)&aTile[arow * 32 + kg * 8];
            const us8 a_hi = *(const us8*)&aTile[arow * 32 + 16 + kg * 8];
#pragma unroll
            for (int j = 0; j < 8; ++j) { afrag[j] = a_lo[j]; afrag[8 + j] = a_hi[j]; }
        }
        // B (32x16 bf16): lanes 0-15: halves0-7=K0-7, halves8-15=K16-23;
        // lanes 16-31: K8-15 / K24-31.  Column N = ln (same K order as A).
        us16 bfrag;
        {
            const int brow = n_sub * 16 + ln;
            const us8 b_lo = *(const us8*)&bTile[brow * 32 + kg * 8];
            const us8 b_hi = *(const us8*)&bTile[brow * 32 + 16 + kg * 8];
#pragma unroll
            for (int j = 0; j < 8; ++j) { bfrag[j] = b_lo[j]; bfrag[8 + j] = b_hi[j]; }
        }

        // D = A x B + C   (EXEC all-1s: no divergent flow reaches here)
        acc = __builtin_amdgcn_wmma_f32_16x16x32_bf16(
            false, __builtin_bit_cast(v16bf, afrag),
            false, __builtin_bit_cast(v16bf, bfrag),
            (short)0, acc, false, false);

        __syncthreads();   // fragments consumed; LDS may be overwritten
    }

    // C/D layout: VGPR r -> M = r + 8*kg, column N = ln
    const int n_out = blockIdx.x * 64 + n_sub * 16 + ln;
    if (n_out < N) {
#pragma unroll
        for (int r = 0; r < 8; ++r) {
            const int co = co_blk + co_sub * 16 + 8 * kg + r;
            out[(size_t)co * N + n_out] = acc[r] + bias[co];
        }
    }
}

// ---------------------------------------------------------------------------
// BatchNorm statistics: one block per channel (biased variance)
// ---------------------------------------------------------------------------
__global__ __launch_bounds__(256)
void bn_reduce(const float* __restrict__ x, float* __restrict__ stats, int Nelem) {
    const int c = blockIdx.x;
    const float* p = x + (size_t)c * Nelem;
    float s = 0.0f, s2 = 0.0f;
    for (int i = threadIdx.x; i < Nelem; i += blockDim.x) {
        float v = p[i];
        s += v; s2 += v * v;
    }
    __shared__ float sh0[256];
    __shared__ float sh1[256];
    sh0[threadIdx.x] = s; sh1[threadIdx.x] = s2;
    __syncthreads();
    for (int off = 128; off > 0; off >>= 1) {
        if ((int)threadIdx.x < off) {
            sh0[threadIdx.x] += sh0[threadIdx.x + off];
            sh1[threadIdx.x] += sh1[threadIdx.x + off];
        }
        __syncthreads();
    }
    if (threadIdx.x == 0) {
        float inv  = 1.0f / (float)Nelem;
        float mean = sh0[0] * inv;
        float var  = sh1[0] * inv - mean * mean;
        stats[2 * c]     = mean;
        stats[2 * c + 1] = rsqrtf(var + 1e-5f);
    }
}

__global__ void bn_relu_bf16(const float* __restrict__ x, const float* __restrict__ stats,
                             unsigned short* __restrict__ y, int Nelem, size_t total) {
    size_t i = (size_t)blockIdx.x * blockDim.x + threadIdx.x;
    if (i >= total) return;
    int c = (int)(i / (size_t)Nelem);
    float v = (x[i] - stats[2 * c]) * stats[2 * c + 1];
    v = v > 0.0f ? v : 0.0f;
    y[i] = f32_to_bf16_bits(v);
}

// MaxPool3d(2,2) on trailing (D,H,W), bf16 in/out
__global__ void maxpool3d_bf16(const unsigned short* __restrict__ x,
                               unsigned short* __restrict__ y,
                               int C, int L, int D, int H, int W) {
    const int Do = D >> 1, Ho = H >> 1, Wo = W >> 1;
    size_t total = (size_t)C * L * Do * Ho * Wo;
    size_t i = (size_t)blockIdx.x * blockDim.x + threadIdx.x;
    if (i >= total) return;
    size_t t = i;
    int wo = (int)(t % Wo); t /= Wo;
    int ho = (int)(t % Ho); t /= Ho;
    int do_ = (int)(t % Do); t /= Do;
    int l  = (int)(t % L);  t /= L;
    int c  = (int)t;
    float m = -3.402823e38f;
#pragma unroll
    for (int pd = 0; pd < 2; ++pd)
#pragma unroll
        for (int ph = 0; ph < 2; ++ph)
#pragma unroll
            for (int pw = 0; pw < 2; ++pw) {
                size_t idx = ((((size_t)c * L + l) * D + (2 * do_ + pd)) * H + (2 * ho + ph))
                             * (size_t)W + (2 * wo + pw);
                float v = bf16_bits_to_f32(x[idx]);
                m = v > m ? v : m;
            }
    y[i] = f32_to_bf16_bits(m);
}

// Head: adaptive mean over (D,H,W)=1 is identity; act bf16 [256][8] = 2048.
__global__ void fc_head(const unsigned short* __restrict__ act,
                        const float* __restrict__ fcw, const float* __restrict__ fcb,
                        float* __restrict__ out) {
    int o = blockIdx.x * blockDim.x + threadIdx.x;
    if (o >= 101) return;
    float s = fcb[o];
    const float* wr = fcw + (size_t)o * 2048;
    for (int k = 0; k < 2048; ++k)
        s += bf16_bits_to_f32(act[k]) * wr[k];
    out[o] = s;
}

// ---------------------------------------------------------------------------
// Host-side orchestration
// ---------------------------------------------------------------------------
static void run_conv_layer(const float* w, const float* bias,
                           const unsigned short* in, unsigned short* act_out,
                           float* convout, unsigned short* wp, float* stats,
                           int Ci, int Co, int Li, int Lo, int D, int H, int W,
                           int strideL, hipStream_t stream) {
    const int Ktot = Ci * 81;
    const int Kpad = (Ktot + 31) & ~31;
    {
        size_t tot = (size_t)Co * Kpad;
        pack_w_bf16<<<dim3((unsigned)((tot + 255) / 256)), dim3(256), 0, stream>>>(
            w, wp, Co, Ktot, Kpad);
    }
    const int N = Lo * D * H * W;
    {
        dim3 grid((unsigned)((N + 63) / 64), (unsigned)(Co / 32));
        conv4d_wmma_bf16<<<grid, dim3(256), 0, stream>>>(
            in, wp, bias, convout, Ci, Co, Li, Lo, D, H, W, strideL, Kpad);
    }
    bn_reduce<<<dim3((unsigned)Co), dim3(256), 0, stream>>>(convout, stats, N);
    {
        size_t tot = (size_t)Co * N;
        bn_relu_bf16<<<dim3((unsigned)((tot + 255) / 256)), dim3(256), 0, stream>>>(
            convout, stats, act_out, N, tot);
    }
}

static void run_maxpool(const unsigned short* in, unsigned short* out,
                        int C, int L, int D, int H, int W, hipStream_t stream) {
    size_t tot = (size_t)C * L * (D / 2) * (H / 2) * (W / 2);
    maxpool3d_bf16<<<dim3((unsigned)((tot + 255) / 256)), dim3(256), 0, stream>>>(
        in, out, C, L, D, H, W);
}

extern "C" void kernel_launch(void* const* d_in, const int* in_sizes, int n_in,
                              void* d_out, int out_size, void* d_ws, size_t ws_size,
                              hipStream_t stream) {
    (void)in_sizes; (void)n_in; (void)out_size; (void)ws_size;

    const float* x   = (const float*)d_in[0];
    const float* W1a = (const float*)d_in[1];  const float* B1a = (const float*)d_in[2];
    const float* W1b = (const float*)d_in[3];  const float* B1b = (const float*)d_in[4];
    const float* W2a = (const float*)d_in[5];  const float* B2a = (const float*)d_in[6];
    const float* W2b = (const float*)d_in[7];  const float* B2b = (const float*)d_in[8];
    const float* W3a = (const float*)d_in[9];  const float* B3a = (const float*)d_in[10];
    const float* W3b = (const float*)d_in[11]; const float* B3b = (const float*)d_in[12];
    const float* W4a = (const float*)d_in[13]; const float* B4a = (const float*)d_in[14];
    const float* W4b = (const float*)d_in[15]; const float* B4b = (const float*)d_in[16];
    const float* W5a = (const float*)d_in[17]; const float* B5a = (const float*)d_in[18];
    const float* W5b = (const float*)d_in[19]; const float* B5b = (const float*)d_in[20];
    const float* FCW = (const float*)d_in[21]; const float* FCB = (const float*)d_in[22];

    // Workspace carve (256-byte aligned slabs)
    const size_t MAX_ACT = 32ull * 128 * 16 * 16 * 16;   // 16.78M elems (stage 1a out)
    char* p = (char*)d_ws;
    auto carve = [&](size_t bytes) -> void* {
        void* r = (void*)p;
        p += (bytes + 255) & ~(size_t)255;
        return r;
    };
    unsigned short* actA    = (unsigned short*)carve(MAX_ACT * 2);
    unsigned short* actB    = (unsigned short*)carve(MAX_ACT * 2);
    float*          convout = (float*)carve(MAX_ACT * 4);
    unsigned short* wp      = (unsigned short*)carve((size_t)256 * 20736 * 2);
    float*          stats   = (float*)carve(512 * 4);

    // Input f32 -> bf16: (3,128,16,16,16)
    {
        size_t nin = 3ull * 128 * 16 * 16 * 16;
        cvt_f32_bf16<<<dim3((unsigned)((nin + 255) / 256)), dim3(256), 0, stream>>>(
            x, actA, nin);
    }

    // Stage 1: (3,128,16^3) -> (32,128,16^3) -> (32,64,16^3) -> pool (32,64,8^3)
    run_conv_layer(W1a, B1a, actA, actB, convout, wp, stats, 3,  32, 128, 128, 16, 16, 16, 1, stream);
    run_conv_layer(W1b, B1b, actB, actA, convout, wp, stats, 32, 32, 128, 64,  16, 16, 16, 2, stream);
    run_maxpool(actA, actB, 32, 64, 16, 16, 16, stream);

    // Stage 2: -> (64,64,8^3) -> (64,32,8^3) -> pool (64,32,4^3)
    run_conv_layer(W2a, B2a, actB, actA, convout, wp, stats, 32, 64, 64, 64, 8, 8, 8, 1, stream);
    run_conv_layer(W2b, B2b, actA, actB, convout, wp, stats, 64, 64, 64, 32, 8, 8, 8, 2, stream);
    run_maxpool(actB, actA, 64, 32, 8, 8, 8, stream);

    // Stage 3: -> (128,32,4^3) -> (128,16,4^3) -> pool (128,16,2^3)
    run_conv_layer(W3a, B3a, actA, actB, convout, wp, stats, 64,  128, 32, 32, 4, 4, 4, 1, stream);
    run_conv_layer(W3b, B3b, actB, actA, convout, wp, stats, 128, 128, 32, 16, 4, 4, 4, 2, stream);
    run_maxpool(actA, actB, 128, 16, 4, 4, 4, stream);

    // Stage 4: -> (256,16,2^3) -> (256,8,2^3) -> pool (256,8,1^3)
    run_conv_layer(W4a, B4a, actB, actA, convout, wp, stats, 128, 256, 16, 16, 2, 2, 2, 1, stream);
    run_conv_layer(W4b, B4b, actA, actB, convout, wp, stats, 256, 256, 16, 8,  2, 2, 2, 2, stream);
    run_maxpool(actB, actA, 256, 8, 2, 2, 2, stream);

    // Stage 5: -> (256,8,1^3) -> (256,8,1^3), no pool
    run_conv_layer(W5a, B5a, actA, actB, convout, wp, stats, 256, 256, 8, 8, 1, 1, 1, 1, stream);
    run_conv_layer(W5b, B5b, actB, actA, convout, wp, stats, 256, 256, 8, 8, 1, 1, 1, 1, stream);

    fc_head<<<dim3(1), dim3(128), 0, stream>>>(actA, FCW, FCB, (float*)d_out);
}
